// EncoderRNN_58256936403339
// MI455X (gfx1250) — compile-verified, matrix operand order checked by reading, not run on hardware
//
#include <hip/hip_runtime.h>

typedef __attribute__((ext_vector_type(16))) __bf16 v16bf;
typedef __attribute__((ext_vector_type(8)))  float  v8f;

constexpr int E    = 512;    // E_DIM
constexpr int H    = 512;    // H_DIM
constexpr int MS   = 300;    // MEM_SLOTS
constexpr int MC   = 300;    // MEM_CTX
constexpr int Bb   = 64;     // batch
constexpr int Tt   = 512;    // seq len
constexpr int MEMP = 320;    // padded mem K (multiple of 32)
constexpr int GD   = 4 * H;  // 2048 gate dim
constexpr int IN   = E + MC; // 812 W_ih inner dim

// tile counts for swizzled operands (each tile = 512 bf16 = one wave fragment)
constexpr int BT    = 4;                       // batch tiles (64/16)
constexpr int KT_IH = 26;                      // (512 + 320)/32 K-tiles of W_ih
constexpr int NT_IH = 128;                     // 2048/16
constexpr int KT_HH = 16;                      // 512/32
constexpr int NT_HH = 128;
constexpr int KT_MK = 16;                      // K = H
constexpr int NT_MK = 19;                      // 304/16 slot tiles
constexpr int KT_MV = 10;                      // K = 320 slot pad
constexpr int NT_MV = 19;                      // 304/16 ctx tiles
constexpr int HT    = 16;                      // h K-tiles (512/32)
constexpr int MT    = 10;                      // mem K-tiles (320/32)

constexpr size_t SZ_C    = (size_t)2 * 2 * Bb * H;         // f32 cell state, ping-pong
constexpr size_t HST     = (size_t)BT * HT * 512;          // bf16 h per (dir,par)
constexpr size_t SZ_H    = 4 * HST;
constexpr size_t MEMST   = (size_t)BT * MT * 512;          // bf16 mem per dir
constexpr size_t SZ_MEM  = 2 * MEMST;
constexpr size_t SZ_WIH  = (size_t)KT_IH * NT_IH * 512;    // per dir
constexpr size_t SZ_WHH  = (size_t)KT_HH * NT_HH * 512;
constexpr size_t SZ_MK   = (size_t)KT_MK * NT_MK * 512;
constexpr size_t SZ_MV   = (size_t)KT_MV * NT_MV * 512;

// ---------- fragment helpers (CDNA5 wave32 layouts, CDNA5 ISA §7.12.2) ----------

__device__ __forceinline__ int frag_kin(int e, int h) {
  int vg = e >> 1, hw = e & 1;
  return ((vg & 3) << 1) + hw + (h << 3) + ((vg >> 2) << 4);
}
__device__ __forceinline__ void frag_inv(int kin, int& h, int& e) {
  h = (kin >> 3) & 1;
  int vg = ((kin >> 4) << 2) | ((kin >> 1) & 3);
  e = (vg << 1) | (kin & 1);
}

// contiguous 32-byte per-lane load from a swizzled bf16 tile array
__device__ __forceinline__ v16bf load_frag_sw(const __bf16* tiles, int tileIdx) {
  int lane = threadIdx.x & 31;
  return *(const v16bf*)(tiles + (size_t)tileIdx * 512 + lane * 16);
}

// unchecked A-fragment from row-major f32 (rows/K provably in-bounds)
__device__ __forceinline__ v16bf load_a_f32(const float* A, int lda, int row0, int k0) {
  int lane = threadIdx.x & 31;
  int half = lane >> 4;
  const float* p = A + (size_t)((lane & 15) + row0) * lda + k0 + (half << 3);
  v16bf out;
#pragma unroll
  for (int e = 0; e < 16; ++e) {
    int vg = e >> 1, hw = e & 1;
    int off = ((vg & 3) << 1) + hw + ((vg >> 2) << 4);  // 0..7 and 16..23: two b128 pairs
    out[e] = (__bf16)p[off];
  }
  return out;
}

__device__ __forceinline__ float sigmf(float x) { return 1.0f / (1.0f + __expf(-x)); }

#define WMMA_BF16(A_, B_, C_) \
  __builtin_amdgcn_wmma_f32_16x16x32_bf16(false, (A_), false, (B_), (short)0, (C_), false, false)

// ---------- one-time weight swizzle: W (N x K row-major f32) -> B^T bf16 fragment tiles ----------
__global__ __launch_bounds__(256) void swizzle_wt_kernel(
    const float* __restrict__ W, int ldw, int Nreal, int Kreal,
    int ntiles, __bf16* __restrict__ outT) {
  int kt = blockIdx.x, nt = blockIdx.y;
  __bf16* dst = outT + ((size_t)kt * ntiles + nt) * 512;
  for (int idx = threadIdx.x; idx < 512; idx += 256) {
    int l = idx >> 4, e = idx & 15;
    int n = nt * 16 + (l & 15);
    int k = kt * 32 + frag_kin(e, l >> 4);
    float v = (n < Nreal && k < Kreal) ? W[(size_t)n * ldw + k] : 0.0f;
    dst[idx] = (__bf16)v;
  }
}

__global__ __launch_bounds__(256) void init_state_kernel(float* c, __bf16* hsw) {
  int i = blockIdx.x * blockDim.x + threadIdx.x;
  if (i < (int)SZ_C) { c[i] = 0.0f; hsw[i] = (__bf16)0.0f; }  // SZ_C == SZ_H
}

// ---------- per step: scores = h@Mk^T, softmax, mem = alpha@Mv^T (stored swizzled bf16) ----------
// grid (4 batch-tiles, 1, 2 dirs), block 256 (8 waves); 3 accumulators per wave share each A-frag
__global__ __launch_bounds__(256) void attn_mem_kernel(
    const __bf16* __restrict__ hsw, int par,
    const __bf16* __restrict__ mksw, const __bf16* __restrict__ mvsw,
    __bf16* __restrict__ memsw, int t) {
  int dir = blockIdx.z;
  int bt  = blockIdx.x;
  int tid = threadIdx.x;
  __bf16* md = memsw + (size_t)dir * MEMST;

  if (t == 0) {  // reference zeroes mem at t==0
    for (int i = tid; i < MT * 512; i += 256) md[(size_t)bt * MT * 512 + i] = (__bf16)0.0f;
    return;
  }

  const __bf16* mk = mksw + (size_t)dir * SZ_MK;
  const __bf16* mv = mvsw + (size_t)dir * SZ_MV;
  const __bf16* h  = hsw + (size_t)(dir * 2 + par) * HST;

  // zero the k in [304,320) padding lanes of mem tile 9 (disjoint from phase-3 writes)
  {
    int m = tid >> 4, kin = 16 + (tid & 15);
    int hh, e; frag_inv(kin, hh, e);
    md[((size_t)bt * MT + 9) * 512 + (m + (hh << 4)) * 16 + e] = (__bf16)0.0f;
  }

  __shared__ float S[16][MEMP];
  __shared__ float red[16][16];
  __shared__ float rowstat[16];
  int wid  = tid >> 5;
  int lane = tid & 31;
  int half = lane >> 4, nn = lane & 15;

  // phase 1: scores (16 x 304); wave owns tiles {w, w+8, w+16(clamped)}; 3 wmma per A-load
  {
    int nt0 = wid, nt1 = wid + 8;
    bool has2 = (wid + 16) < NT_MK;
    int nt2 = has2 ? (wid + 16) : (NT_MK - 1);
    v8f a0 = {}, a1 = {}, a2 = {};
#pragma unroll 2
    for (int kt = 0; kt < KT_MK; ++kt) {
      v16bf a  = load_frag_sw(h, bt * HT + kt);
      v16bf b0 = load_frag_sw(mk, kt * NT_MK + nt0);
      v16bf b1 = load_frag_sw(mk, kt * NT_MK + nt1);
      v16bf b2 = load_frag_sw(mk, kt * NT_MK + nt2);
      a0 = WMMA_BF16(a, b0, a0);
      a1 = WMMA_BF16(a, b1, a1);
      a2 = WMMA_BF16(a, b2, a2);
    }
#pragma unroll
    for (int r = 0; r < 8; ++r) {
      S[r + (half << 3)][nn + nt0 * 16] = a0[r];
      S[r + (half << 3)][nn + nt1 * 16] = a1[r];
    }
    if (has2) {
#pragma unroll
      for (int r = 0; r < 8; ++r) S[r + (half << 3)][nn + nt2 * 16] = a2[r];
    }
  }
  __syncthreads();

  // phase 2: row softmax over 300 valid cols; pad cols forced to exact 0 weight
  int r = tid >> 4, sub = tid & 15;
  float mx = -3.0e38f;
  for (int c = sub; c < MS; c += 16) mx = fmaxf(mx, S[r][c]);
  red[r][sub] = mx;
  __syncthreads();
  if (sub == 0) {
    float m = red[r][0];
    for (int j = 1; j < 16; ++j) m = fmaxf(m, red[r][j]);
    rowstat[r] = m;
  }
  __syncthreads();
  float rmax = rowstat[r];
  float sm = 0.0f;
  for (int c = sub; c < MEMP; c += 16) {
    float ev = (c < MS) ? __expf(S[r][c] - rmax) : 0.0f;
    S[r][c] = ev;
    sm += ev;
  }
  red[r][sub] = sm;
  __syncthreads();
  if (sub == 0) {
    float s = 0.0f;
    for (int j = 0; j < 16; ++j) s += red[r][j];
    rowstat[r] = 1.0f / s;
  }
  __syncthreads();
  float inv = rowstat[r];
  for (int c = sub; c < MS; c += 16) S[r][c] *= inv;
  __syncthreads();

  // phase 3: mem = alpha @ Mv^T; 3 accs per wave; D scattered into swizzled bf16 A-layout
  {
    int nt0 = wid, nt1 = wid + 8;
    bool has2 = (wid + 16) < NT_MV;
    int nt2 = has2 ? (wid + 16) : (NT_MV - 1);
    v8f a0 = {}, a1 = {}, a2 = {};
#pragma unroll 2
    for (int kt = 0; kt < KT_MV; ++kt) {
      v16bf a  = load_a_f32(&S[0][0], MEMP, 0, kt * 32);
      v16bf b0 = load_frag_sw(mv, kt * NT_MV + nt0);
      v16bf b1 = load_frag_sw(mv, kt * NT_MV + nt1);
      v16bf b2 = load_frag_sw(mv, kt * NT_MV + nt2);
      a0 = WMMA_BF16(a, b0, a0);
      a1 = WMMA_BF16(a, b1, a1);
      a2 = WMMA_BF16(a, b2, a2);
    }
#pragma unroll
    for (int q = 0; q < 3; ++q) {
      if (q == 2 && !has2) break;
      int nt = (q == 0) ? nt0 : (q == 1) ? nt1 : nt2;
      v8f acc = (q == 0) ? a0 : (q == 1) ? a1 : a2;
      int n = nn + nt * 16;              // ctx col (K of gates GEMM)
      int kt2 = n >> 5, kin = n & 31, hh, e;
      frag_inv(kin, hh, e);
      __bf16* dtile = md + ((size_t)bt * MT + kt2) * 512;
#pragma unroll
      for (int r2 = 0; r2 < 8; ++r2) {
        int m = r2 + (half << 3);
        dtile[(m + (hh << 4)) * 16 + e] = (__bf16)acc[r2];
      }
    }
  }
}

// ---------- per step: gates + LSTM update; one wave owns all 4 gates for its 16 hidden cols ----------
// grid (4 batch-tiles, 8 col-groups, 2 dirs), block 128 (4 waves, wave w -> n0 = y*64 + w*16)
// 4 accumulators (i,f,g,o) per wave: 4 wmma per shared A-frag, pointwise update fully in registers.
__global__ __launch_bounds__(128) void gates_lstm_kernel(
    const float* __restrict__ x, const long long* __restrict__ lens,
    const __bf16* __restrict__ wihsw, const __bf16* __restrict__ whhsw,
    const float* __restrict__ bih_f, const float* __restrict__ bih_b,
    const float* __restrict__ bhh_f, const float* __restrict__ bhh_b,
    __bf16* __restrict__ hsw, float* __restrict__ cbase,
    const __bf16* __restrict__ memsw, int par, int t, float* __restrict__ out) {
  int dir  = blockIdx.z;
  int bt   = blockIdx.x;
  int b0   = bt * 16;
  int wid  = threadIdx.x >> 5;
  int lane = threadIdx.x & 31;
  int n0   = blockIdx.y * 64 + wid * 16;  // hidden column base, 0..496
  int nct  = n0 >> 4;                     // hidden n-tile 0..31

  const __bf16* wih = wihsw + (size_t)dir * SZ_WIH;
  const __bf16* whh = whhsw + (size_t)dir * SZ_WHH;
  const float*  bih = dir ? bih_b : bih_f;
  const float*  bhh = dir ? bhh_b : bhh_f;
  const __bf16* h_in  = hsw + (size_t)(dir * 2 + par) * HST;
  __bf16*       h_out = hsw + (size_t)(dir * 2 + (par ^ 1)) * HST;
  const float*  c_in  = cbase + (size_t)(dir * 2 + par) * Bb * H;
  float*        c_out = cbase + (size_t)(dir * 2 + (par ^ 1)) * Bb * H;
  const __bf16* mem   = memsw + (size_t)dir * MEMST;
  int torig = dir ? (Tt - 1 - t) : t;

  int half = lane >> 4, nn = lane & 15;
  int hc = n0 + nn;  // this lane's hidden column

  v8f ai, af, ag, ao;
  {
    float bi = bih[hc] + bhh[hc];
    float bf_ = bih[H + hc] + bhh[H + hc];
    float bg = bih[2 * H + hc] + bhh[2 * H + hc];
    float bo = bih[3 * H + hc] + bhh[3 * H + hc];
#pragma unroll
    for (int q = 0; q < 8; ++q) { ai[q] = bi; af[q] = bf_; ag[q] = bg; ao[q] = bo; }
  }

  // x contribution (f32 A, in-bounds -> clean b128 + cvt_pk); B col-tiles g*32 + nct
  const float* A0 = x + (size_t)torig * E;  // row stride T*E over batch
#pragma unroll 2
  for (int kt = 0; kt < 16; ++kt) {
    v16bf a  = load_a_f32(A0, Tt * E, b0, kt * 32);
    v16bf bi_ = load_frag_sw(wih, kt * NT_IH + nct);
    v16bf bf2 = load_frag_sw(wih, kt * NT_IH + 32 + nct);
    v16bf bg_ = load_frag_sw(wih, kt * NT_IH + 64 + nct);
    v16bf bo_ = load_frag_sw(wih, kt * NT_IH + 96 + nct);
    ai = WMMA_BF16(a, bi_, ai);
    af = WMMA_BF16(a, bf2, af);
    ag = WMMA_BF16(a, bg_, ag);
    ao = WMMA_BF16(a, bo_, ao);
  }
  // mem contribution (K-tiles 16..25 of W_ih)
#pragma unroll 2
  for (int kt = 16; kt < 26; ++kt) {
    v16bf a  = load_frag_sw(mem, bt * MT + (kt - 16));
    v16bf bi_ = load_frag_sw(wih, kt * NT_IH + nct);
    v16bf bf2 = load_frag_sw(wih, kt * NT_IH + 32 + nct);
    v16bf bg_ = load_frag_sw(wih, kt * NT_IH + 64 + nct);
    v16bf bo_ = load_frag_sw(wih, kt * NT_IH + 96 + nct);
    ai = WMMA_BF16(a, bi_, ai);
    af = WMMA_BF16(a, bf2, af);
    ag = WMMA_BF16(a, bg_, ag);
    ao = WMMA_BF16(a, bo_, ao);
  }
  // recurrent contribution
#pragma unroll 2
  for (int kt = 0; kt < 16; ++kt) {
    v16bf a  = load_frag_sw(h_in, bt * HT + kt);
    v16bf bi_ = load_frag_sw(whh, kt * NT_HH + nct);
    v16bf bf2 = load_frag_sw(whh, kt * NT_HH + 32 + nct);
    v16bf bg_ = load_frag_sw(whh, kt * NT_HH + 64 + nct);
    v16bf bo_ = load_frag_sw(whh, kt * NT_HH + 96 + nct);
    ai = WMMA_BF16(a, bi_, ai);
    af = WMMA_BF16(a, bf2, af);
    ag = WMMA_BF16(a, bg_, ag);
    ao = WMMA_BF16(a, bo_, ao);
  }

  // LSTM pointwise update fully in registers (acc element r -> row m = r + 8*half, col hc)
  int ktile = hc >> 5, kin = hc & 31, hh, e;
  frag_inv(kin, hh, e);
  __bf16* hdst = h_out + ((size_t)bt * HT + ktile) * 512;
#pragma unroll
  for (int r = 0; r < 8; ++r) {
    int m = r + (half << 3);
    int b = b0 + m;
    float iv = sigmf(ai[r]);
    float fv = sigmf(af[r]);
    float gv = tanhf(ag[r]);
    float ov = sigmf(ao[r]);
    float cold = c_in[(size_t)b * H + hc];
    float cnew = fv * cold + iv * gv;
    float hnew = ov * tanhf(cnew);
    c_out[(size_t)b * H + hc] = cnew;
    hdst[(m + (hh << 4)) * 16 + e] = (__bf16)hnew;
    float o = (torig < (int)lens[b]) ? hnew : 0.0f;
    out[((size_t)b * Tt + torig) * (size_t)(2 * H) + (size_t)dir * H + hc] = o;
  }
}

extern "C" void kernel_launch(void* const* d_in, const int* in_sizes, int n_in,
                              void* d_out, int out_size, void* d_ws, size_t ws_size,
                              hipStream_t stream) {
  const float*     x     = (const float*)d_in[0];
  const long long* lens  = (const long long*)d_in[1];
  const float*     Wih_f = (const float*)d_in[2];
  const float*     Whh_f = (const float*)d_in[3];
  const float*     bih_f = (const float*)d_in[4];
  const float*     bhh_f = (const float*)d_in[5];
  const float*     Mk_f  = (const float*)d_in[6];
  const float*     Mv_f  = (const float*)d_in[7];
  const float*     Wih_b = (const float*)d_in[8];
  const float*     Whh_b = (const float*)d_in[9];
  const float*     bih_b = (const float*)d_in[10];
  const float*     bhh_b = (const float*)d_in[11];
  const float*     Mk_b  = (const float*)d_in[12];
  const float*     Mv_b  = (const float*)d_in[13];
  float* out = (float*)d_out;

  // workspace carve-up (f32 region first, then 32B-aligned bf16 regions; ~13 MB, L2-resident)
  float*  cA   = (float*)d_ws;
  __bf16* bf   = (__bf16*)(cA + SZ_C);
  __bf16* hsw  = bf;               bf += SZ_H;
  __bf16* msw  = bf;               bf += SZ_MEM;
  __bf16* wih  = bf;               bf += 2 * SZ_WIH;
  __bf16* whh  = bf;               bf += 2 * SZ_WHH;
  __bf16* mks  = bf;               bf += 2 * SZ_MK;
  __bf16* mvs  = bf;               bf += 2 * SZ_MV;

  // one-time per call: zero state, swizzle weights to bf16 fragment tiles
  init_state_kernel<<<dim3((unsigned)((SZ_C + 255) / 256)), 256, 0, stream>>>(cA, hsw);
  swizzle_wt_kernel<<<dim3(KT_IH, NT_IH), 256, 0, stream>>>(Wih_f, IN, GD, IN, NT_IH, wih);
  swizzle_wt_kernel<<<dim3(KT_IH, NT_IH), 256, 0, stream>>>(Wih_b, IN, GD, IN, NT_IH, wih + SZ_WIH);
  swizzle_wt_kernel<<<dim3(KT_HH, NT_HH), 256, 0, stream>>>(Whh_f, H, GD, H, NT_HH, whh);
  swizzle_wt_kernel<<<dim3(KT_HH, NT_HH), 256, 0, stream>>>(Whh_b, H, GD, H, NT_HH, whh + SZ_WHH);
  swizzle_wt_kernel<<<dim3(KT_MK, NT_MK), 256, 0, stream>>>(Mk_f, H, MS, H, NT_MK, mks);
  swizzle_wt_kernel<<<dim3(KT_MK, NT_MK), 256, 0, stream>>>(Mk_b, H, MS, H, NT_MK, mks + SZ_MK);
  swizzle_wt_kernel<<<dim3(KT_MV, NT_MV), 256, 0, stream>>>(Mv_f, MS, MC, MS, NT_MV, mvs);
  swizzle_wt_kernel<<<dim3(KT_MV, NT_MV), 256, 0, stream>>>(Mv_b, MS, MC, MS, NT_MV, mvs + SZ_MV);

  for (int t = 0; t < Tt; ++t) {
    int par = t & 1;
    attn_mem_kernel<<<dim3(BT, 1, 2), 256, 0, stream>>>(hsw, par, mks, mvs, msw, t);
    gates_lstm_kernel<<<dim3(BT, 8, 2), 128, 0, stream>>>(
        x, lens, wih, whh, bih_f, bih_b, bhh_f, bhh_b, hsw, cA, msw, par, t, out);
  }
}
